// VectorQuantizer_8203387536014
// MI455X (gfx1250) — matrix operand order repaired and checked by reading, not compiled, not run on hardware
//
#include <hip/hip_runtime.h>

typedef __attribute__((ext_vector_type(16))) _Float16 v16h;
typedef __attribute__((ext_vector_type(8)))  float    v8f;

#define NUM_CODES 4096
#define DIM       64
#define NROWS     65536            /* 16 * 4096 */
#define NGROUPS   (NUM_CODES / 16)
#define COMMIT    0.25f

union V16H { v16h v; _Float16 h[16]; };
union V8F  { v8f  v; float    f[8];  };

// ---------------------------------------------------------------------------
// Kernel 1: convert codebook W (fp32) -> f16, precompute ||w_k||^2
// ---------------------------------------------------------------------------
__global__ __launch_bounds__(256) void vq_prep_w(const float* __restrict__ W,
                                                 _Float16* __restrict__ W16,
                                                 float* __restrict__ wnorm) {
    int k = blockIdx.x * blockDim.x + threadIdx.x;
    if (k >= NUM_CODES) return;
    const float* wr = W + (size_t)k * DIM;
    _Float16*    wo = W16 + (size_t)k * DIM;
    float s = 0.0f;
#pragma unroll
    for (int j = 0; j < DIM; ++j) {
        float v = wr[j];
        wo[j] = (_Float16)v;
        s += v * v;
    }
    wnorm[k] = s;
}

// ---------------------------------------------------------------------------
// Kernel 2: per-row argmin over 4096 codes via v_wmma_f32_16x16x32_f16.
// Each wave: 32 rows (two 16-row M tiles, A resident in VGPRs), streams all
// codes in groups of 16.  dist = ||w||^2 - 2 * (z . w)  (||z||^2 dropped).
// ---------------------------------------------------------------------------
__global__ __launch_bounds__(256) void vq_argmin(const float* __restrict__ z,
                                                 const _Float16* __restrict__ W16,
                                                 const float* __restrict__ wnorm,
                                                 int* __restrict__ argidx) {
    const int lane = threadIdx.x & 31;
    const int hi   = lane >> 4;       // which 16-lane half
    const int lr   = lane & 15;
    const int wave  = blockIdx.x * (blockDim.x >> 5) + (threadIdx.x >> 5);
    const int rbase = wave * 32;      // 32 rows per wave

    // ---- Load + convert A tiles (z rows) once; dense 16-bit A layout:
    // half lo: K 0..15; half hi: K 16..31 (per 32-chunk), matching B.
    V16H a[2][2];
#pragma unroll
    for (int t = 0; t < 2; ++t) {
        const float* zrow = z + (size_t)(rbase + t * 16 + lr) * DIM;
#pragma unroll
        for (int c = 0; c < 2; ++c) {
            const int kc = c * 32;
            const int o0 = kc + hi * 16;
            const int o1 = kc + hi * 16 + 8;
#pragma unroll
            for (int e = 0; e < 8; ++e) a[t][c].h[e]     = (_Float16)zrow[o0 + e];
#pragma unroll
            for (int e = 0; e < 8; ++e) a[t][c].h[8 + e] = (_Float16)zrow[o1 + e];
        }
    }

    float bestD[2][8];
    int   bestI[2][8];
#pragma unroll
    for (int t = 0; t < 2; ++t)
#pragma unroll
        for (int r = 0; r < 8; ++r) { bestD[t][r] = 3.4e38f; bestI[t][r] = 0; }

    for (int g = 0; g < NGROUPS; ++g) {
        const int cbase = g * 16;
        const int code  = cbase + lr;              // this lane's column (N)
        // B layout (32x16, 16-bit): half lo holds K 0..15, half hi K 16..31,
        // column N = lane%16  ->  one contiguous 32B load per K-chunk.
        const _Float16* wrow = W16 + (size_t)code * DIM + hi * 16;
        v16h b0 = *(const v16h*)(wrow);            // K  0..31 chunk
        v16h b1 = *(const v16h*)(wrow + 32);       // K 32..63 chunk
        float wn = wnorm[code];

        if (g + 1 < NGROUPS) {                     // global_prefetch_b8 next group
            __builtin_prefetch(W16 + (size_t)(cbase + 16) * DIM + lane * 32, 0, 1);
        }

#pragma unroll
        for (int t = 0; t < 2; ++t) {
            V8F acc = {};
            acc.v = __builtin_amdgcn_wmma_f32_16x16x32_f16(
                        false, a[t][0].v, false, b0, (short)0, acc.v, false, false);
            acc.v = __builtin_amdgcn_wmma_f32_16x16x32_f16(
                        false, a[t][1].v, false, b1, (short)0, acc.v, false, false);
#pragma unroll
            for (int r = 0; r < 8; ++r) {
                float d = wn - 2.0f * acc.f[r];
                if (d < bestD[t][r]) { bestD[t][r] = d; bestI[t][r] = code; }
            }
        }
    }

    // Cross-lane argmin within each 16-lane half (C/D: N = lane%16,
    // rows: half lo -> M r, half hi -> M 8+r). First-min tie-break on index.
#pragma unroll
    for (int t = 0; t < 2; ++t) {
#pragma unroll
        for (int r = 0; r < 8; ++r) {
            float d = bestD[t][r];
            int   i = bestI[t][r];
#pragma unroll
            for (int m = 1; m < 16; m <<= 1) {
                float od = __shfl_xor(d, m, 32);
                int   oi = __shfl_xor(i, m, 32);
                if (od < d || (od == d && oi < i)) { d = od; i = oi; }
            }
            if (lr == 0) argidx[rbase + t * 16 + hi * 8 + r] = i;
        }
    }
}

// ---------------------------------------------------------------------------
// Kernel 3: gather z_q from fp32 codebook, write straight-through output,
// accumulate per-block partial sums of (z - z_q)^2 (deterministic order).
// ---------------------------------------------------------------------------
__global__ __launch_bounds__(256) void vq_gather(const float* __restrict__ z,
                                                 const float* __restrict__ W,
                                                 const int* __restrict__ argidx,
                                                 float* __restrict__ out,
                                                 float* __restrict__ partials) {
    __shared__ float sred[256];
    const size_t total  = (size_t)NROWS * DIM;
    const size_t stride = (size_t)gridDim.x * blockDim.x;
    float local = 0.0f;
    for (size_t e = (size_t)blockIdx.x * blockDim.x + threadIdx.x; e < total; e += stride) {
        size_t row = e >> 6;
        int    j   = (int)(e & 63);
        float zq = W[(size_t)argidx[row] * DIM + j];
        float zv = z[e];
        out[1 + e] = zv + (zq - zv);       // straight-through estimator
        float df = zv - zq;
        local += df * df;
    }
    sred[threadIdx.x] = local;
    __syncthreads();
    for (int s = 128; s > 0; s >>= 1) {
        if ((int)threadIdx.x < s) sred[threadIdx.x] += sred[threadIdx.x + s];
        __syncthreads();
    }
    if (threadIdx.x == 0) partials[blockIdx.x] = sred[0];
}

// ---------------------------------------------------------------------------
// Kernel 4: reduce 256 partials, scale, write loss to out[0].
// ---------------------------------------------------------------------------
__global__ __launch_bounds__(256) void vq_finalize(const float* __restrict__ partials,
                                                   float* __restrict__ out) {
    __shared__ float sred[256];
    sred[threadIdx.x] = partials[threadIdx.x];
    __syncthreads();
    for (int s = 128; s > 0; s >>= 1) {
        if ((int)threadIdx.x < s) sred[threadIdx.x] += sred[threadIdx.x + s];
        __syncthreads();
    }
    if (threadIdx.x == 0)
        out[0] = sred[0] * (COMMIT / ((float)NROWS * (float)DIM));
}

// ---------------------------------------------------------------------------
extern "C" void kernel_launch(void* const* d_in, const int* in_sizes, int n_in,
                              void* d_out, int out_size, void* d_ws, size_t ws_size,
                              hipStream_t stream) {
    const float* z = (const float*)d_in[0];   // [16,4096,64] fp32
    const float* W = (const float*)d_in[1];   // [4096,64]    fp32
    float* out = (float*)d_out;               // [1 + 16*4096*64]

    char* ws = (char*)d_ws;
    _Float16* W16     = (_Float16*)ws;                                   // 512 KB
    float*    wnorm   = (float*)(ws + 524288);                           //  16 KB
    int*      argidx  = (int*)  (ws + 524288 + 16384);                   // 256 KB
    float*    partials= (float*)(ws + 524288 + 16384 + 262144);          //   1 KB

    vq_prep_w <<<NUM_CODES / 256, 256, 0, stream>>>(W, W16, wnorm);
    vq_argmin <<<(NROWS / 32) / 8, 256, 0, stream>>>(z, W16, wnorm, argidx); // 256 blocks, 8 waves each
    vq_gather <<<256, 256, 0, stream>>>(z, W, argidx, out, partials);
    vq_finalize<<<1, 256, 0, stream>>>(partials, out);
}
